// sinc_conv_10256381903617
// MI455X (gfx1250) — matrix operand order repaired and checked by reading, not compiled, not run on hardware
//
#include <hip/hip_runtime.h>
#include <hip/hip_bf16.h>

// ---------------------------------------------------------------------------
// SincNet conv for MI455X (gfx1250, wave32, WMMA + Tensor Data Mover).
//   out[b,f,t] = sum_k W[f,k] * x[b, 8*t + k],  f<40, k<101, t<19988
// GEMM form: W (40x101 -> padded 48x128 f16) x patches (128 x 19988).
// Roofline: 20.7 GFLOP vs 409 MB output -> output-bandwidth bound.
//   - TDM (tensor_load_to_lds) stages each block's input window into LDS
//     with zero VALU/VGPR traffic; tensor_dim OOB gives free zero-padding.
//   - WMMA f32<-f16 does the math (trivial vs matrix-core peak).
//   - Non-temporal stores stream the 409 MB output past the 192 MB L2.
// ---------------------------------------------------------------------------

typedef __attribute__((ext_vector_type(16))) _Float16 v16h;
typedef __attribute__((ext_vector_type(8)))  _Float16 v8h;
typedef __attribute__((ext_vector_type(8)))  float    v8f;
typedef __attribute__((ext_vector_type(4)))  unsigned int v4u;
typedef __attribute__((ext_vector_type(8)))  int      v8i;
typedef __attribute__((ext_vector_type(4)))  int      v4i;

#define FS_F     16000.0f
#define N_FILT   40
#define FILT_DIM 101
#define M_PAD    48        // 3 WMMA M-tiles of 16
#define K_PAD    128       // 4 WMMA K-chunks of 32
#define T_IN     160000
#define BATCH    128
#define STRIDE   8
#define T_OUT    19988     // (160000 - 101)/8 + 1
#define NBLK     256       // output time-steps per block (8 waves x 2 n-tiles x 16)
#define SPAN     (NBLK * STRIDE + K_PAD)   // 2176 staged input samples (f32, 8.5 KB LDS)
#define PI_F     3.14159265358979323846f

// low_pass(fc)[k] of the reference (fc = normalized freq): arg = 2*pi*fc*j.
__device__ __forceinline__ float sinc_lp(float fc, int k) {
  if (k == 50) return 2.0f * fc;                  // center tap
  int j = (k > 50) ? (k - 50) : (50 - k);         // mirrored right half, j = 1..50
  float arg = 2.0f * PI_F * fc * (float)j;
  return 2.0f * fc * (sinf(arg) / arg);
}

// ---------------------------------------------------------------------------
// Kernel 1: build padded f16 filter matrix W[M_PAD][K_PAD] in workspace.
// ---------------------------------------------------------------------------
__global__ void build_filters_kernel(const float* __restrict__ filt_b1,
                                     const float* __restrict__ filt_band,
                                     _Float16* __restrict__ wf16) {
  const int tid = threadIdx.x;
  for (int i = tid; i < M_PAD * K_PAD; i += blockDim.x)
    wf16[i] = (_Float16)0.0f;
  __syncthreads();

  if (tid < N_FILT) {
    const float min_f = 50.0f / FS_F;
    float beg = fabsf(filt_b1[tid]) + min_f;
    float end = beg + fabsf(filt_band[tid]) + min_f;

    float maxv = -3.0e38f;
    for (int k = 0; k < FILT_DIM; ++k) {
      float bp = sinc_lp(end, k) - sinc_lp(beg, k);
      maxv = fmaxf(maxv, bp);
    }
    const float inv = 1.0f / maxv;
    for (int k = 0; k < FILT_DIM; ++k) {
      float bp  = sinc_lp(end, k) - sinc_lp(beg, k);
      float win = 0.54f - 0.46f * cosf(2.0f * PI_F * (float)k / 100.0f);
      wf16[tid * K_PAD + k] = (_Float16)(bp * inv * win);
    }
  }
}

// ---------------------------------------------------------------------------
// Kernel 2: TDM-staged, WMMA-tiled GEMM. Block = 256 thr (8 waves),
// block tile 48 filters x 256 times, wave tile 48 x 32.
// ---------------------------------------------------------------------------
__global__ void __launch_bounds__(256)
sinc_conv_wmma_kernel(const float* __restrict__ x,
                      const _Float16* __restrict__ wf16,
                      float* __restrict__ out) {
  __shared__ float lds[SPAN];                       // f32 staging (TDM target)

  const int b   = blockIdx.y;
  const int t0  = blockIdx.x * NBLK;
  const int tid = threadIdx.x;
  const float* __restrict__ xb = x + (size_t)b * T_IN;
  const int gbase = t0 * STRIDE;

  // ---- Tensor Data Mover: DMA this block's input window into LDS. ----
  // 1-row tile of SPAN f32. tensor_dim0 = elements remaining from the tile
  // start, so TDM's OOB rule (reads past tensor_dim return 0) implements the
  // signal-tail zero fill. Issued by wave 0 only; tracked with TENSORcnt.
  if (tid < 32) {
    const unsigned long long gaddr = (unsigned long long)(uintptr_t)(xb + gbase);
    const unsigned int lds_off = (unsigned int)(uintptr_t)(&lds[0]); // low32 = LDS byte offset
    const unsigned int rem = (unsigned int)(T_IN - gbase);           // gbase < T_IN always

    v4u g0;
    g0[0] = 1u;                                            // count=1, user D#
    g0[1] = lds_off;                                       // lds_addr (bytes)
    g0[2] = (unsigned int)(gaddr & 0xFFFFFFFFu);           // global_addr[31:0]
    g0[3] = (unsigned int)((gaddr >> 32) & 0x01FFFFFFu)    // global_addr[56:32]
            | (2u << 30);                                  // type = 2 ("image")

    v8i g1;
    g1[0] = (int)(2u << 16);                               // wg_mask=0, data_size=4B
    g1[1] = (int)((rem & 0xFFFFu) << 16);                  // tensor_dim0[15:0] @ bits63:48
    g1[2] = (int)((rem >> 16) | (1u << 16));               // tensor_dim0[31:16]; tensor_dim1=1
    g1[3] = (int)((unsigned int)SPAN << 16);               // tile_dim0 @ bits127:112
    g1[4] = 1;                                             // tile_dim1=1, tile_dim2=0
    g1[5] = (int)rem;                                      // tensor_dim0_stride[31:0]
    g1[6] = 0;                                             // stride hi, dim1_stride lo
    g1[7] = 0;
    v4i g2 = {0, 0, 0, 0};                                 // <=2-D tensor
    v4i g3 = {0, 0, 0, 0};
    v8i g4 = {0, 0, 0, 0, 0, 0, 0, 0};                     // trailing group (clang-23 form)
    __builtin_amdgcn_tensor_load_to_lds(g0, g1, g2, g3, g4, 0);
  }
  __builtin_amdgcn_s_wait_tensorcnt(0);   // no-op for non-issuing waves
  __syncthreads();

  const int wave = tid >> 5;     // 0..7
  const int lane = tid & 31;
  const int col  = lane & 15;    // N column / A row within tile
  const int hi   = lane >> 4;    // lane-group selector for K split

  v8f acc[3][2];
#pragma unroll
  for (int mt = 0; mt < 3; ++mt)
#pragma unroll
    for (int nt = 0; nt < 2; ++nt)
      acc[mt][nt] = (v8f)(0.0f);

  const int kb_a = hi * 16;      // A k order used here (contiguous, same as B): lanes16-31 hold K=16..31
  const int kb_b = hi * 16;      // B k order used here (contiguous, same as A): lanes16-31 hold K=16..31

#pragma unroll
  for (int kc = 0; kc < 4; ++kc) {
    // --- A fragments (filters, f16 in global; L2-resident) ---
    v16h afr[3];
#pragma unroll
    for (int mt = 0; mt < 3; ++mt) {
      const _Float16* p = wf16 + (mt * 16 + col) * K_PAD + kc * 32 + kb_a;
      v8h lo = *(const v8h*)(p);        // K = kb_a + 0..7   (16B aligned)
      v8h hh = *(const v8h*)(p + 8);    // K = kb_a + 8..15  (16B aligned)
#pragma unroll
      for (int e = 0; e < 8; ++e) { afr[mt][e] = lo[e]; afr[mt][e + 8] = hh[e]; }
    }
    // --- B fragments: f32 from LDS, convert to f16 (v_cvt pairs) ---
    v16h bfr[2];
#pragma unroll
    for (int nt = 0; nt < 2; ++nt) {
      const int n_local = wave * 32 + nt * 16 + col;
      const float* q = lds + n_local * STRIDE + kc * 32 + kb_b;  // 32B aligned
      v8f f0 = *(const v8f*)(q);
      v8f f1 = *(const v8f*)(q + 8);
#pragma unroll
      for (int e = 0; e < 8; ++e) {
        bfr[nt][e]     = (_Float16)f0[e];
        bfr[nt][e + 8] = (_Float16)f1[e];
      }
    }
    // --- 6 WMMAs per k-chunk ---
#pragma unroll
    for (int mt = 0; mt < 3; ++mt)
#pragma unroll
      for (int nt = 0; nt < 2; ++nt)
        acc[mt][nt] = __builtin_amdgcn_wmma_f32_16x16x32_f16(
            false, afr[mt], false, bfr[nt], (short)0, acc[mt][nt],
            false, false);
  }

  // --- Store: C/D layout -> out[b][f][t]; stream with non-temporal stores ---
#pragma unroll
  for (int mt = 0; mt < 3; ++mt) {
#pragma unroll
    for (int nt = 0; nt < 2; ++nt) {
      const int t = t0 + wave * 32 + nt * 16 + col;
      if (t < T_OUT) {
#pragma unroll
        for (int r = 0; r < 8; ++r) {
          const int f = mt * 16 + r + hi * 8;   // C row: M = r + 8*(lane>=16)
          if (f < N_FILT) {
            float* dst = out + ((size_t)b * N_FILT + f) * T_OUT + t;
            __builtin_nontemporal_store(acc[mt][nt][r], dst);
          }
        }
      }
    }
  }
}

// ---------------------------------------------------------------------------
extern "C" void kernel_launch(void* const* d_in, const int* in_sizes, int n_in,
                              void* d_out, int out_size, void* d_ws, size_t ws_size,
                              hipStream_t stream) {
  (void)in_sizes; (void)n_in; (void)out_size; (void)ws_size;
  const float* x     = (const float*)d_in[0];
  const float* fb1   = (const float*)d_in[1];
  const float* fband = (const float*)d_in[2];
  float*       out   = (float*)d_out;
  _Float16*    wf16  = (_Float16*)d_ws;   // 48*128 f16 = 12 KB of workspace

  build_filters_kernel<<<dim3(1), dim3(256), 0, stream>>>(fb1, fband, wf16);

  dim3 grid((T_OUT + NBLK - 1) / NBLK, BATCH);   // 79 x 128
  sinc_conv_wmma_kernel<<<grid, dim3(256), 0, stream>>>(x, wf16, out);
}
